// Net_2224793059748
// MI455X (gfx1250) — matrix operand order repaired and checked by reading, not compiled, not run on hardware
//
#include <hip/hip_runtime.h>
#include <hip/hip_bf16.h>
#include <math.h>

// ---------------------------------------------------------------------------
// CDNA5 (gfx1250) wave32 WMMA bf16 helpers
// ---------------------------------------------------------------------------
typedef __attribute__((ext_vector_type(16))) __bf16    v16bf;
typedef __attribute__((ext_vector_type(8)))  float     v8f;
typedef __attribute__((ext_vector_type(4)))  unsigned  v4u;

union ABFrag { v16bf v; v4u q[2]; };

// A-matrix 16x32 bf16 fragment from LDS, row-major [M][ldk].
// ISA 7.12.2: lane half h, row m=lane&15; K = 2v+p+8h (v<4), 16+2(v-4)+p+8h:
// two contiguous 16B runs at elements 8h and 16+8h.
__device__ __forceinline__ v16bf load_a_frag(const __bf16* lds, int row_base,
                                             int k_base, int ldk, int lane) {
    const int half = lane >> 4, m = lane & 15;
    const __bf16* p = lds + (size_t)(row_base + m) * ldk + k_base + 8 * half;
    ABFrag f;
    f.q[0] = *(const v4u*)(p);        // K = 8h + {0..7}
    f.q[1] = *(const v4u*)(p + 16);   // K = 16 + 8h + {0..7}
    return f.v;
}

// B-matrix 32x16 bf16 fragment. B[k][n] stored as [N][ldk] row-major
// (W row-major [out][in] is already this layout for X @ W^T).
// lane half h, N=lane&15: K = 8h+{0..7} and 16+8h+{0..7} (same order as A).
__device__ __forceinline__ v16bf load_b_frag(const __bf16* lds, int n_base,
                                             int k_base, int ldk, int lane) {
    const int half = lane >> 4, n = lane & 15;
    const __bf16* p = lds + (size_t)(n_base + n) * ldk + k_base + 8 * half;
    ABFrag f;
    f.q[0] = *(const v4u*)(p);
    f.q[1] = *(const v4u*)(p + 16);
    return f.v;
}

__device__ __forceinline__ v8f wmma_bf16(v16bf a, v16bf b, v8f c) {
    return __builtin_amdgcn_wmma_f32_16x16x32_bf16(false, a, false, b,
                                                   (short)0, c, false, false);
}

// Software-pipelined K-chain: B fragment for kk+1 is issued BEFORE the WMMA
// for kk, so each LDS fetch overlaps a matrix op (wait dscnt<=2, not 0).
template <int NK>
__device__ __forceinline__ v8f gemm_chain(const v16bf* a, const __bf16* Blds,
                                          int n_base, int ldk, int lane) {
    v8f c = {};
    v16bf bcur = load_b_frag(Blds, n_base, 0, ldk, lane);
#pragma unroll
    for (int kk = 0; kk < NK; ++kk) {
        v16bf bnxt = bcur;
        if (kk + 1 < NK)
            bnxt = load_b_frag(Blds, n_base, (kk + 1) * 32, ldk, lane);
        c = wmma_bf16(a[kk], bcur, c);
        bcur = bnxt;
    }
    return c;
}

// ---------------------------------------------------------------------------
// Problem dimensions
// ---------------------------------------------------------------------------
#define SEQ_LEN 200
#define BATCH   1024
#define EMB     100
#define HID     300
#define FC1N    256
#define NCLS    3

#define HPN 304   // HIDDEN padded to 16 (N-tile dimension)
#define HPK 320   // HIDDEN padded to 32 (K-step dimension)
#define EPK 128   // EMB padded to 32

// ---------------------------------------------------------------------------
// Phase 1: xp[s*B+b][n] = emb(x[s,b]) @ W_ih^T + b_ih + b_hh   (f32 out)
// 1600 blocks x 256 thr (8 waves); 128 rows x 304 cols per block.
// ---------------------------------------------------------------------------
__global__ __launch_bounds__(256) void k_embed_inproj(
    const int* __restrict__ x, const float* __restrict__ W_em,
    const float* __restrict__ W_ih, const float* __restrict__ b_ih,
    const float* __restrict__ b_hh, float* __restrict__ xp) {
    extern __shared__ char smem[];
    __bf16* A     = (__bf16*)smem;                           // [128][128]
    __bf16* Bm    = (__bf16*)(smem + 128 * EPK * 2);         // [304][128]
    float*  biasL = (float*)(smem + (128 + HPN) * EPK * 2);  // [304]
    const int tid  = threadIdx.x;
    const int row0 = blockIdx.x * 128;

    // B = W_ih [300][100] -> [304][128] bf16, zero padded
    for (int i = tid; i < HPN * EPK; i += 256) {
        int n = i >> 7, k = i & 127;
        float v = (n < HID && k < EMB) ? W_ih[n * EMB + k] : 0.f;
        Bm[i] = (__bf16)v;
    }
    // A = gathered embedding rows, zero padded in K
    for (int i = tid; i < 128 * EPK; i += 256) {
        int r = i >> 7, k = i & 127;
        int tok = x[row0 + r];
        float v = (k < EMB) ? W_em[(size_t)tok * EMB + k] : 0.f;
        A[i] = (__bf16)v;
    }
    // combined bias, staged once (kills per-tile predicated global loads)
    for (int i = tid; i < HPN; i += 256)
        biasL[i] = (i < HID) ? (b_ih[i] + b_hh[i]) : 0.f;
    __syncthreads();

    const int wave = tid >> 5, lane = tid & 31;
    const int m_base = wave * 16;
    const int nlane = lane & 15, half = lane >> 4;

    v16bf a[4];
#pragma unroll
    for (int kk = 0; kk < 4; ++kk)
        a[kk] = load_a_frag(A, m_base, kk * 32, EPK, lane);

    for (int nt = 0; nt < 19; ++nt) {
        const int n_base = nt * 16;
        const int n = n_base + nlane;
        const float bias = biasL[n];  // early ds load, consumed after chain
        v8f c = gemm_chain<4>(a, Bm, n_base, EPK, lane);
#pragma unroll
        for (int r = 0; r < 8; ++r) {
            int grow = row0 + m_base + r + 8 * half;
            xp[(size_t)grow * HPN + n] = c[r] + bias;
        }
    }
}

// ---------------------------------------------------------------------------
// Phase 2: 200-step recurrence h' = tanh(xp[t] + h @ W_hh^T)
// W_hh lives in LDS (bf16, 190KB) for the whole time loop; h double-buffered.
// 16 blocks x 64 batch rows, 8 waves: wave owns M-tile (w&3), N-parity (w>>2).
// ---------------------------------------------------------------------------
__global__ __launch_bounds__(256) void k_rnn(
    const float* __restrict__ W_hh, const float* __restrict__ xp,
    float* __restrict__ hfin) {
    extern __shared__ char smem[];
    __bf16* Whh   = (__bf16*)smem;                        // [304][320]
    __bf16* hbuf0 = (__bf16*)(smem + HPN * HPK * 2);      // [64][320]
    __bf16* hbuf1 = hbuf0 + 64 * HPK;                     // [64][320]
    const int tid  = threadIdx.x;
    const int brow = blockIdx.x * 64;

    for (int i = tid; i < HPN * HPK; i += 256) {
        int n = i / HPK, k = i % HPK;
        float v = (n < HID && k < HID) ? W_hh[n * HID + k] : 0.f;
        Whh[i] = (__bf16)v;
    }
    for (int i = tid; i < 2 * 64 * HPK; i += 256) hbuf0[i] = (__bf16)0.f;
    __syncthreads();

    const int wave = tid >> 5, lane = tid & 31;
    const int m_base = (wave & 3) * 16;
    const int npar = wave >> 2;
    const int nlane = lane & 15, half = lane >> 4;

    // hot loop: t = 0 .. 198 (final step peeled: stores f32 to global)
    for (int t = 0; t < SEQ_LEN - 1; ++t) {
        __bf16* cur = (t & 1) ? hbuf1 : hbuf0;
        __bf16* nxt = (t & 1) ? hbuf0 : hbuf1;

        v16bf a[10];
#pragma unroll
        for (int kk = 0; kk < 10; ++kk)
            a[kk] = load_a_frag(cur, m_base, kk * 32, HPK, lane);

        for (int nt = npar; nt < 19; nt += 2) {
            const int n_base = nt * 16;
            const int n = n_base + nlane;

            // xp loads issued early; they drain under the WMMA chain
            const float* xrow =
                xp + ((size_t)t * BATCH + brow + m_base + 8 * half) * HPN + n;
            float xv[8];
#pragma unroll
            for (int r = 0; r < 8; ++r) xv[r] = xrow[(size_t)r * HPN];
            // prefetch next timestep's slice of xp into cache
            __builtin_prefetch((const void*)(xrow + (size_t)BATCH * HPN), 0, 1);

            v8f c = gemm_chain<10>(a, Whh, n_base, HPK, lane);

#pragma unroll
            for (int r = 0; r < 8; ++r) {
                float hv = tanhf(c[r] + xv[r]);
                int lrow = m_base + r + 8 * half;
                nxt[(size_t)lrow * HPK + n] = (__bf16)hv;
            }
        }
        __syncthreads();  // writes to nxt visible before next step reads it
    }

    // final step t = SEQ_LEN-1: result goes to global memory in f32
    {
        const int t = SEQ_LEN - 1;
        __bf16* cur = (t & 1) ? hbuf1 : hbuf0;
        v16bf a[10];
#pragma unroll
        for (int kk = 0; kk < 10; ++kk)
            a[kk] = load_a_frag(cur, m_base, kk * 32, HPK, lane);

        for (int nt = npar; nt < 19; nt += 2) {
            const int n_base = nt * 16;
            const int n = n_base + nlane;
            const float* xrow =
                xp + ((size_t)t * BATCH + brow + m_base + 8 * half) * HPN + n;
            float xv[8];
#pragma unroll
            for (int r = 0; r < 8; ++r) xv[r] = xrow[(size_t)r * HPN];

            v8f c = gemm_chain<10>(a, Whh, n_base, HPK, lane);
#pragma unroll
            for (int r = 0; r < 8; ++r) {
                int grow = brow + m_base + r + 8 * half;
                hfin[(size_t)grow * HPN + n] = tanhf(c[r] + xv[r]);
            }
        }
    }
}

// ---------------------------------------------------------------------------
// Phase 3a: z = relu(hfin @ W1^T + b1)   M=1024 K=300(pad 320) N=256
// ---------------------------------------------------------------------------
__global__ __launch_bounds__(256) void k_fc1(
    const float* __restrict__ hfin, const float* __restrict__ W1,
    const float* __restrict__ b1, float* __restrict__ z) {
    extern __shared__ char smem[];
    __bf16* A     = (__bf16*)smem;                          // [64][320]
    __bf16* Bm    = (__bf16*)(smem + 64 * HPK * 2);         // [256][320]
    float*  biasL = (float*)(smem + (64 + FC1N) * HPK * 2); // [256]
    const int tid  = threadIdx.x;
    const int row0 = blockIdx.x * 64;

    for (int i = tid; i < FC1N * HPK; i += 256) {
        int n = i / HPK, k = i % HPK;
        float v = (k < HID) ? W1[n * HID + k] : 0.f;
        Bm[i] = (__bf16)v;
    }
    for (int i = tid; i < 64 * HPK; i += 256) {
        int r = i / HPK, k = i % HPK;
        float v = (k < HPN) ? hfin[(size_t)(row0 + r) * HPN + k] : 0.f;
        A[i] = (__bf16)v;
    }
    if (tid < FC1N) biasL[tid] = b1[tid];
    __syncthreads();

    const int wave = tid >> 5, lane = tid & 31;
    const int m_base = (wave & 3) * 16;
    const int npar = wave >> 2;
    const int nlane = lane & 15, half = lane >> 4;

    v16bf a[10];
#pragma unroll
    for (int kk = 0; kk < 10; ++kk)
        a[kk] = load_a_frag(A, m_base, kk * 32, HPK, lane);

    for (int nt = npar; nt < 16; nt += 2) {
        const int n_base = nt * 16;
        const int n = n_base + nlane;
        const float bias = biasL[n];
        v8f c = gemm_chain<10>(a, Bm, n_base, HPK, lane);
#pragma unroll
        for (int r = 0; r < 8; ++r) {
            float v = c[r] + bias;
            v = v > 0.f ? v : 0.f;
            z[(size_t)(row0 + m_base + r + 8 * half) * FC1N + n] = v;
        }
    }
}

// ---------------------------------------------------------------------------
// Phase 3b: out = z @ W2^T + b2   (N=3: too narrow for a 16x16 tile -> VALU)
// ---------------------------------------------------------------------------
__global__ __launch_bounds__(256) void k_fc2(
    const float* __restrict__ z, const float* __restrict__ W2,
    const float* __restrict__ b2, float* __restrict__ out) {
    int i = blockIdx.x * 256 + threadIdx.x;
    if (i >= BATCH * NCLS) return;
    int row = i / NCLS, cls = i % NCLS;
    const float* zr = z + (size_t)row * FC1N;
    const float* wr = W2 + cls * FC1N;
    float acc = b2[cls];
#pragma unroll 8
    for (int k = 0; k < FC1N; ++k) acc += zr[k] * wr[k];
    out[i] = acc;
}

// ---------------------------------------------------------------------------
extern "C" void kernel_launch(void* const* d_in, const int* in_sizes, int n_in,
                              void* d_out, int out_size, void* d_ws,
                              size_t ws_size, hipStream_t stream) {
    (void)in_sizes; (void)n_in; (void)out_size; (void)ws_size;
    const int*   x    = (const int*)d_in[0];
    const float* W_em = (const float*)d_in[1];
    const float* W_ih = (const float*)d_in[2];
    const float* W_hh = (const float*)d_in[3];
    const float* b_ih = (const float*)d_in[4];
    const float* b_hh = (const float*)d_in[5];
    const float* W1   = (const float*)d_in[6];
    const float* b1   = (const float*)d_in[7];
    const float* W2   = (const float*)d_in[8];
    const float* b2   = (const float*)d_in[9];

    char* ws = (char*)d_ws;
    float* xp   = (float*)ws;                                         // S*B*304 f32
    float* hfin = (float*)(ws + (size_t)SEQ_LEN * BATCH * HPN * 4);   // B*304 f32
    float* z    = hfin + (size_t)BATCH * HPN;                         // B*256 f32

    // Phase 1: LDS = (128+304)*128*2 + 304*4 = 111808 B
    k_embed_inproj<<<(SEQ_LEN * BATCH) / 128, 256, 111808, stream>>>(
        x, W_em, W_ih, b_ih, b_hh, xp);

    // Phase 2: LDS = 304*320*2 + 2*64*320*2 = 276480 B
    k_rnn<<<BATCH / 64, 256, 276480, stream>>>(W_hh, xp, hfin);

    // Phase 3a: LDS = (64+256)*320*2 + 256*4 = 205824 B
    k_fc1<<<BATCH / 64, 256, 205824, stream>>>(hfin, W1, b1, z);

    // Phase 3b
    k_fc2<<<(BATCH * NCLS + 255) / 256, 256, 0, stream>>>(z, W2, b2,
                                                          (float*)d_out);
}
